// TrainableTFNLayer2D_82248623718879
// MI455X (gfx1250) — matrix operand order repaired and checked by reading, not compiled, not run on hardware
//
#include <hip/hip_runtime.h>
#include <hip/hip_bf16.h>
#include <stdint.h>

typedef _Float16 h16;
typedef __attribute__((ext_vector_type(16))) _Float16 v16h;
typedef __attribute__((ext_vector_type(8)))  _Float16 v8h;
typedef __attribute__((ext_vector_type(8)))  float    v8f;
typedef __attribute__((ext_vector_type(2)))  float    v2f;
typedef __attribute__((ext_vector_type(4)))  unsigned int v4u;
typedef __attribute__((ext_vector_type(8)))  int      v8i;
typedef __attribute__((ext_vector_type(4)))  int      v4i;

#define MB (1u << 20)
#define KB (1u << 10)

#if __has_builtin(__builtin_amdgcn_tensor_load_to_lds)
#define TFN_TDM 1
#else
#define TFN_TDM 0
#endif

// ---------------- WMMA wrappers ----------------
__device__ __forceinline__ v8f wmma16(v16h a, v16h b, v8f c) {
  return __builtin_amdgcn_wmma_f32_16x16x32_f16(false, a, false, b, (short)0, c, false, false);
}
__device__ __forceinline__ v8f wmma4(v2f a, v2f b, v8f c) {
  return __builtin_amdgcn_wmma_f32_16x16x4_f32(false, a, false, b, (short)0, c, false, false);
}

// A-fragment (16-bit, 16xK): lane (m=lane&15, hi=lane>>4); element i maps to
// K offset (i<8?0:16) + hi*8 + (i&7)  => two contiguous 16B chunks.
__device__ __forceinline__ v16h fragA(const h16* row, int hi) {
  v8h lo = *(const v8h*)(row + hi * 8);
  v8h hp = *(const v8h*)(row + 16 + hi * 8);
  return __builtin_shufflevector(lo, hp, 0, 1, 2, 3, 4, 5, 6, 7, 8, 9, 10, 11, 12, 13, 14, 15);
}
// B-fragment: element i maps to K = (i<8?0:16) + hi*8 + (i&7) => two contiguous 16B chunks of BT row n.
__device__ __forceinline__ v16h fragB(const h16* row, int hi) {
  v8h lo = *(const v8h*)(row + hi * 8);
  v8h hp = *(const v8h*)(row + 16 + hi * 8);
  return __builtin_shufflevector(lo, hp, 0, 1, 2, 3, 4, 5, 6, 7, 8, 9, 10, 11, 12, 13, 14, 15);
}

#if TFN_TDM
// Issue one 1-D TDM load: 4096 contiguous f16 (one NHWC field row) -> LDS.
// Descriptor layout per CDNA5 ISA 8.3/8.4 (D# group0 128b, group1 256b; 2-D groups 2/3 zero).
__device__ __forceinline__ void tdm_load_row(unsigned lds_byte, const h16* gsrc) {
  unsigned long long ga = (unsigned long long)(uintptr_t)gsrc;
  v4u g0;
  g0[0] = 1u;                                               // count=1, user mode
  g0[1] = lds_byte;                                         // lds_addr (bytes)
  g0[2] = (unsigned)(ga & 0xffffffffu);                     // global_addr[31:0]
  g0[3] = (unsigned)((ga >> 32) & 0x1ffffffu) | (2u << 30); // global_addr[56:32], type=2
  v8i g1;
  g1[0] = (int)(1u << 16);          // workgroup_mask=0 (not in cluster), data_size=1 (2B)
  g1[1] = (int)(4096u << 16);       // tensor_dim0[15:0]=4096
  g1[2] = (int)(0u | (1u << 16));   // tensor_dim0[31:16]=0, tensor_dim1[15:0]=1
  g1[3] = (int)(4096u << 16);       // tensor_dim1[31:16]=0, tile_dim0=4096
  g1[4] = (int)1;                   // tile_dim1=1, tile_dim2=0
  g1[5] = (int)4096;                // tensor_dim0_stride[31:0]
  g1[6] = 0;                        // stride0 hi, stride1 lo
  g1[7] = 0;                        // stride1 hi
  v4i g2 = {};
  v4i g3 = {};
#if __clang_major__ >= 23
  v8i g4 = {};
  __builtin_amdgcn_tensor_load_to_lds(g0, g1, g2, g3, g4, 0);
#else
  __builtin_amdgcn_tensor_load_to_lds(g0, g1, g2, g3, 0);
#endif
}
__device__ __forceinline__ void tdm_wait0() {
#if __has_builtin(__builtin_amdgcn_s_wait_tensorcnt)
  __builtin_amdgcn_s_wait_tensorcnt((short)0);
#else
  asm volatile("s_wait_tensorcnt 0x0" ::: "memory");
#endif
}
#endif

// ---------------- problem sizes ----------------
#define EE 512
#define DD 64
#define HH 64
#define WW 64
#define BB 8
#define LL 2048

// ---------------- kernel 0: weight repack (f32 -> f16, GEMM-friendly layouts) ----------------
// w1T[co][tap*64+ci] (128x576), w2T[co][ci] (64x128)
__global__ void tfn_prep(const float* __restrict__ c1w, const float* __restrict__ c2w,
                         h16* __restrict__ w1T, h16* __restrict__ w2T) {
  int i = blockIdx.x * 256 + threadIdx.x;
  if (i < 128 * 576) {
    int co = i / 576, k = i % 576;
    int tap = k >> 6, ci = k & 63;
    int ty = tap / 3, tx = tap % 3;
    w1T[i] = (h16)c1w[((co * 64 + ci) * 3 + ty) * 3 + tx];
  } else if (i < 128 * 576 + 64 * 128) {
    int j = i - 128 * 576;
    w2T[j] = (h16)c2w[j];
  }
}

// ---------------- kernel 1: separable Gaussians ----------------
__global__ void tfn_gauss(const float* __restrict__ positions, const float* __restrict__ log_sigma,
                          h16* __restrict__ gyT, h16* __restrict__ gxT,
                          h16* __restrict__ gyoT, h16* __restrict__ gxo,
                          float* __restrict__ invn) {
  int tid = blockIdx.x * 128 + threadIdx.x;  // 16384
  int b = tid >> 11, l = tid & 2047;
  float py = positions[tid * 2 + 0];
  float px = positions[tid * 2 + 1];
  float ls = log_sigma[0];
  float s_in = log1pf(expf(ls)) + 1e-6f;
  float s_out = s_in * 2.0f;
  float ki = 1.0f / (2.0f * s_in * s_in);
  float ko = 1.0f / (2.0f * s_out * s_out);
  float syi = 0.f, sxi = 0.f, syo = 0.f, sxo = 0.f;
  for (int h = 0; h < 64; ++h) {
    float dy = (float)h - py, dx = (float)h - px;
    float dy2 = dy * dy, dx2 = dx * dx;
    syi += expf(-dy2 * ki); sxi += expf(-dx2 * ki);
    syo += expf(-dy2 * ko); sxo += expf(-dx2 * ko);
  }
  float inv_i = 1.0f / (syi * sxi + 1e-6f);
  float inv_o = 1.0f / (syo * sxo + 1e-6f);
  invn[tid] = inv_i;
  for (int h = 0; h < 64; ++h) {
    float dy = (float)h - py, dx = (float)h - px;
    float dy2 = dy * dy, dx2 = dx * dx;
    gyT[(b * 64 + h) * LL + l]  = (h16)expf(-dy2 * ki);
    gxT[(b * 64 + h) * LL + l]  = (h16)expf(-dx2 * ki);
    gyoT[(b * 64 + h) * LL + l] = (h16)(expf(-dy2 * ko) * inv_o);
    gxo[tid * 64 + h]           = (h16)expf(-dx2 * ko);
  }
}

// ---------------- kernel 2: proj = tokens @ W_to_field^T, scaled by invn -> projnT (b,d,L) f16 ----
__global__ __launch_bounds__(128) void tfn_proj(const float* __restrict__ tokens,
                                                const float* __restrict__ w2f,
                                                const float* __restrict__ invn,
                                                h16* __restrict__ projnT) {
  int wave = blockIdx.x * 4 + (threadIdx.x >> 5);
  int lane = threadIdx.x & 31, n16 = lane & 15, hi = lane >> 4;
  int m0 = wave * 16;
  v8f acc[4] = {};
  const float* arow = tokens + (size_t)(m0 + n16) * EE;
  for (int e0 = 0; e0 < EE; e0 += 4) {
    if (e0 + 128 < EE) __builtin_prefetch(arow + e0 + 128, 0, 0);
    v2f a = *(const v2f*)(arow + e0 + hi * 2);
#pragma unroll
    for (int nt = 0; nt < 4; ++nt) {
      v2f bv = *(const v2f*)(w2f + (size_t)(nt * 16 + n16) * EE + e0 + hi * 2);
      acc[nt] = wmma4(a, bv, acc[nt]);
    }
  }
#pragma unroll
  for (int r = 0; r < 8; ++r) {
    int bl = m0 + r + hi * 8;
    float iv = invn[bl];
    int b = bl >> 11, l = bl & 2047;
#pragma unroll
    for (int nt = 0; nt < 4; ++nt) {
      int d = nt * 16 + n16;
      projnT[(size_t)(b * 64 + d) * LL + l] = (h16)(acc[nt][r] * iv);
    }
  }
}

// ---------------- kernel 3: splat  field[b,h,w,c] = sum_l projn[l,c]*gy[l,h]*gx[l,w] -------------
__global__ __launch_bounds__(128) void tfn_splat(const h16* __restrict__ projnT,
                                                 const h16* __restrict__ gyT,
                                                 const h16* __restrict__ gxT,
                                                 float* __restrict__ f32o, h16* __restrict__ f16o) {
  int b = blockIdx.x >> 6, h = blockIdx.x & 63;
  int wave = threadIdx.x >> 5, lane = threadIdx.x & 31;
  int n16 = lane & 15, hi = lane >> 4;
  int w0 = wave * 16;
  const h16* gyrow = gyT + (size_t)(b * 64 + h) * LL;
  const h16* gxrow = gxT + (size_t)(b * 64 + w0 + n16) * LL;
  const h16* ar[4];
#pragma unroll
  for (int mt = 0; mt < 4; ++mt) ar[mt] = projnT + (size_t)(b * 64 + mt * 16 + n16) * LL;
  v8f acc[4] = {};
  for (int l0 = 0; l0 < LL; l0 += 32) {
    if (l0 + 128 < LL) __builtin_prefetch(gxrow + l0 + 128, 0, 0);
    v16h gy = fragA(gyrow + l0, hi);
    v16h bf = fragB(gxrow + l0, hi);
#pragma unroll
    for (int mt = 0; mt < 4; ++mt) {
      v16h a = fragA(ar[mt] + l0, hi) * gy;
      acc[mt] = wmma16(a, bf, acc[mt]);
    }
  }
#pragma unroll
  for (int mt = 0; mt < 4; ++mt)
#pragma unroll
    for (int r = 0; r < 8; ++r) {
      int c = mt * 16 + r + hi * 8;
      int w = w0 + n16;
      size_t idx = ((size_t)(b * 64 + h) * 64 + w) * 64 + c;
      f32o[idx] = acc[mt][r];
      f16o[idx] = (h16)acc[mt][r];
    }
}

// ---------------- kernel 4: one residual conv step (implicit GEMM, NHWC f16, f32 residual) -------
// LDS row tile: [gut64][row0 4096][gut64][row1 4096][gut64][row2 4096][gut64]; pitch 4160 halves.
// Each interior row is one contiguous 8KB transfer -> Tensor Data Mover when available.
__global__ __launch_bounds__(128) void tfn_conv(const float* __restrict__ in32,
                                                const h16* __restrict__ in16,
                                                float* __restrict__ out32, h16* __restrict__ out16,
                                                h16* __restrict__ out16T, int last,
                                                const h16* __restrict__ w1T, const float* __restrict__ b1,
                                                const h16* __restrict__ w2T, const float* __restrict__ b2) {
  __shared__ h16 xin[3 * 4160 + 64];
  __shared__ h16 hbuf[64 * 128];
  int b = blockIdx.x >> 6, h = blockIdx.x & 63;
  int tid = threadIdx.x;
  // zero the 4 halo gutters (64 halves each at r*4160)
  if (tid < 32) {
    int r = tid >> 3, q = tid & 7;
    v8h z = {};
    *(v8h*)(xin + r * 4160 + q * 8) = z;
  }
#if TFN_TDM
  {
    bool v0 = (h > 0), v2 = (h < 63);
    if (tid < 32) {  // wave 0 only (wave-uniform branch); TDM ignores EXEC
      unsigned ldsbase = (unsigned)(unsigned long long)(uintptr_t)xin;
      if (v0) tdm_load_row(ldsbase + (0 * 4160 + 64) * 2, in16 + (size_t)(b * 64 + h - 1) * 4096);
      tdm_load_row(ldsbase + (1 * 4160 + 64) * 2, in16 + (size_t)(b * 64 + h) * 4096);
      if (v2) tdm_load_row(ldsbase + (2 * 4160 + 64) * 2, in16 + (size_t)(b * 64 + h + 1) * 4096);
    }
    v8h z = {};
    if (!v0) for (int q = tid; q < 512; q += 128) *(v8h*)(xin + 0 * 4160 + 64 + q * 8) = z;
    if (!v2) for (int q = tid; q < 512; q += 128) *(v8h*)(xin + 2 * 4160 + 64 + q * 8) = z;
    tdm_wait0();  // no-op for waves 1..3 (their TENSORcnt==0)
  }
#else
  for (int r3 = 0; r3 < 3; ++r3) {
    int hh = h - 1 + r3;
    h16* dst = xin + r3 * 4160 + 64;
    if (hh >= 0 && hh < 64) {
      const h16* src = in16 + (size_t)(b * 64 + hh) * 4096;
      for (int q = tid; q < 512; q += 128) *(v8h*)(dst + q * 8) = *(const v8h*)(src + q * 8);
    } else {
      v8h z = {};
      for (int q = tid; q < 512; q += 128) *(v8h*)(dst + q * 8) = z;
    }
  }
#endif
  __syncthreads();

  int wave = tid >> 5, lane = tid & 31, n16 = lane & 15, hi = lane >> 4;
  int nb = wave * 32;                 // conv1 N-chunk per wave
  v8f acc[2][4] = {};
  for (int kb = 0; kb < 18; ++kb) {   // K = 9 taps * 64 ci, step 32
    int tap = kb >> 1, ci0 = (kb & 1) * 32;
    int r3 = tap / 3, dx = tap % 3;
    v16h bf0 = fragB(w1T + (size_t)(nb + n16) * 576 + kb * 32, hi);
    v16h bf1 = fragB(w1T + (size_t)(nb + 16 + n16) * 576 + kb * 32, hi);
#pragma unroll
    for (int mt = 0; mt < 4; ++mt) {
      const h16* p = xin + r3 * 4160 + (mt * 16 + n16 + dx) * 64 + ci0;
      v16h a = fragA(p, hi);
      acc[0][mt] = wmma16(a, bf0, acc[0][mt]);
      acc[1][mt] = wmma16(a, bf1, acc[1][mt]);
    }
  }
  float bias0 = b1[nb + n16], bias1 = b1[nb + 16 + n16];
#pragma unroll
  for (int nt = 0; nt < 2; ++nt)
#pragma unroll
    for (int mt = 0; mt < 4; ++mt)
#pragma unroll
      for (int r = 0; r < 8; ++r) {
        int m = mt * 16 + r + hi * 8;
        int co = nb + nt * 16 + n16;
        float v = acc[nt][mt][r] + (nt ? bias1 : bias0);
        hbuf[m * 128 + co] = (h16)(v > 0.f ? v : 0.f);
      }
  __syncthreads();

  int co0 = wave * 16;                // conv2 N-tile per wave
  v8f a2[4] = {};
  for (int kb = 0; kb < 4; ++kb) {
    v16h bf = fragB(w2T + (size_t)(co0 + n16) * 128 + kb * 32, hi);
#pragma unroll
    for (int mt = 0; mt < 4; ++mt) {
      v16h a = fragA(hbuf + (mt * 16 + n16) * 128 + kb * 32, hi);
      a2[mt] = wmma16(a, bf, a2[mt]);
    }
  }
  float bb = b2[co0 + n16];
#pragma unroll
  for (int mt = 0; mt < 4; ++mt)
#pragma unroll
    for (int r = 0; r < 8; ++r) {
      int w = mt * 16 + r + hi * 8;
      int c = co0 + n16;
      size_t idx = ((size_t)(b * 64 + h) * 64 + w) * 64 + c;
      float v = in32[idx] + a2[mt][r] + bb;
      if (!last) {
        out32[idx] = v;
        out16[idx] = (h16)v;
      } else {
        out16T[(size_t)(b * 64 + c) * 4096 + h * 64 + w] = (h16)v;  // (b, c, hw) for the gather
      }
    }
}

// ---------------- kernel 5: gather  sampled[b,l,c] = sum_hw gyo*gxo*field[c,hw] ----------------
__global__ __launch_bounds__(128) void tfn_sample(const h16* __restrict__ gyoT,
                                                  const h16* __restrict__ gxo,
                                                  const h16* __restrict__ f16T,
                                                  float* __restrict__ sampled) {
  int b = blockIdx.x >> 5, lt = blockIdx.x & 31;
  int wave = threadIdx.x >> 5, lane = threadIdx.x & 31;
  int n16 = lane & 15, hi = lane >> 4;
  int lrow = lt * 64 + wave * 16 + n16;
  const h16* gxrow = gxo + ((size_t)b * LL + lrow) * 64;
  v8f acc[4] = {};
  for (int kb = 0; kb < 128; ++kb) {
    int h = kb >> 1, wk0 = (kb & 1) * 32;
    if (kb + 8 < 128) __builtin_prefetch(f16T + (size_t)(b * 64 + n16) * 4096 + (kb + 8) * 32, 0, 0);
    _Float16 gy = gyoT[(size_t)(b * 64 + h) * LL + lrow];
    v16h a = fragA(gxrow + wk0, hi) * gy;
#pragma unroll
    for (int nt = 0; nt < 4; ++nt) {
      v16h bf = fragB(f16T + (size_t)(b * 64 + nt * 16 + n16) * 4096 + kb * 32, hi);
      acc[nt] = wmma16(a, bf, acc[nt]);
    }
  }
#pragma unroll
  for (int nt = 0; nt < 4; ++nt)
#pragma unroll
    for (int r = 0; r < 8; ++r) {
      int l = lt * 64 + wave * 16 + r + hi * 8;
      int c = nt * 16 + n16;
      sampled[((size_t)b * LL + l) * 64 + c] = acc[nt][r];
    }
}

// ---------------- kernel 6: tokens_out = sampled @ W_from_field^T (f32 WMMA) ----------------
__global__ __launch_bounds__(128) void tfn_out(const float* __restrict__ sampled,
                                               const float* __restrict__ wff,
                                               float* __restrict__ out) {
  int m0 = blockIdx.x * 16;
  int wave = threadIdx.x >> 5, lane = threadIdx.x & 31;
  int n16 = lane & 15, hi = lane >> 4;
  int nbase = wave * 128;
  v8f acc[8] = {};
  const float* arow = sampled + (size_t)(m0 + n16) * 64;
  for (int d0 = 0; d0 < 64; d0 += 4) {
    v2f a = *(const v2f*)(arow + d0 + hi * 2);
#pragma unroll
    for (int nt = 0; nt < 8; ++nt) {
      v2f bv = *(const v2f*)(wff + (size_t)(nbase + nt * 16 + n16) * 64 + d0 + hi * 2);
      acc[nt] = wmma4(a, bv, acc[nt]);
    }
  }
#pragma unroll
  for (int nt = 0; nt < 8; ++nt)
#pragma unroll
    for (int r = 0; r < 8; ++r) {
      int m = m0 + r + hi * 8;
      int e = nbase + nt * 16 + n16;
      out[(size_t)m * EE + e] = acc[nt][r];
    }
}

// ---------------- launch ----------------
extern "C" void kernel_launch(void* const* d_in, const int* in_sizes, int n_in,
                              void* d_out, int out_size, void* d_ws, size_t ws_size,
                              hipStream_t stream) {
  const float* tokens    = (const float*)d_in[0];
  const float* positions = (const float*)d_in[1];
  const float* w2f       = (const float*)d_in[2];
  const float* wff       = (const float*)d_in[3];
  const float* c1w       = (const float*)d_in[4];
  const float* c1b       = (const float*)d_in[5];
  const float* c2w       = (const float*)d_in[6];
  const float* c2b       = (const float*)d_in[7];
  const float* lsig      = (const float*)d_in[8];
  float* out = (float*)d_out;

  char* ws = (char*)d_ws;
  h16*   projnT  = (h16*)(ws + 0);                    // 2 MB (b, d, L)
  h16*   gyT     = (h16*)(ws + 2 * MB);               // 2 MB (b, h, L)
  h16*   gxT     = (h16*)(ws + 4 * MB);               // 2 MB (b, w, L)
  h16*   gyoT    = (h16*)(ws + 6 * MB);               // 2 MB (b, h, L) * inv_norm_out
  h16*   gxo     = (h16*)(ws + 8 * MB);               // 2 MB (b, L, w)
  float* invn    = (float*)(ws + 10 * MB);            // 64 KB
  h16*   w1T     = (h16*)(ws + 10 * MB + 128 * KB);   // 144 KB (co, tap*64+ci)
  h16*   w2T     = (h16*)(ws + 10 * MB + 512 * KB);   // 16 KB  (co, ci)
  float* sampled = (float*)(ws + 11 * MB);            // 4 MB (b, L, d)
  float* f32A    = (float*)(ws + 16 * MB);            // 8 MB (b, h, w, c)
  float* f32B    = (float*)(ws + 24 * MB);            // 8 MB
  h16*   f16A    = (h16*)(ws + 32 * MB);              // 4 MB
  h16*   f16B    = (h16*)(ws + 36 * MB);              // 4 MB
  h16*   f16T    = (h16*)(ws + 40 * MB);              // 4 MB (b, c, hw)

  tfn_prep<<<320, 256, 0, stream>>>(c1w, c2w, w1T, w2T);
  tfn_gauss<<<128, 128, 0, stream>>>(positions, lsig, gyT, gxT, gyoT, gxo, invn);
  tfn_proj<<<256, 128, 0, stream>>>(tokens, w2f, invn, projnT);
  tfn_splat<<<512, 128, 0, stream>>>(projnT, gyT, gxT, f32A, f16A);
  for (int s = 0; s < 5; ++s) {
    int last = (s == 4);
    const float* i32 = (s & 1) ? f32B : f32A;
    const h16*   i16 = (s & 1) ? f16B : f16A;
    float* o32 = (s & 1) ? f32A : f32B;
    h16*   o16 = (s & 1) ? f16A : f16B;
    tfn_conv<<<512, 128, 0, stream>>>(i32, i16, o32, o16, f16T, last, w1T, c1b, w2T, c2b);
  }
  tfn_sample<<<256, 128, 0, stream>>>(gyoT, gxo, f16T, sampled);
  tfn_out<<<1024, 128, 0, stream>>>(sampled, wff, out);
}
